// GausskeyL2P_29394756174149
// MI455X (gfx1250) — matrix operand order repaired and checked by reading, not compiled, not run on hardware
//
#include <hip/hip_runtime.h>
#include <hip/hip_bf16.h>
#include <math.h>

// ---------------------------------------------------------------------------
// GausskeyL2P forward for MI455X (gfx1250, wave32, WMMA).
// All large GEMMs go through v_wmma_f32_16x16x32_bf16 (f32->bf16 on load).
// Gaussian log-prob via blocked Cholesky + forward substitution (no inverse).
// ---------------------------------------------------------------------------

#define BDIM 768
#define NHEAD 12
#define DHEAD 64
#define NB_ 32
#define S1 197
#define S2 222
#define POOLN 10
#define SELN 5
#define PLENN 5

#define FLAG_BT   1
#define FLAG_GELU 2

typedef __attribute__((ext_vector_type(16))) __bf16 v16bf;
typedef __attribute__((ext_vector_type(8)))  float  v8f;
typedef __attribute__((ext_vector_type(4)))  unsigned int v4u;

union fragu { v16bf v; v4u q[2]; };

__device__ __forceinline__ unsigned int pack_bf16(float lo, float hi) {
  union { __bf16 h[2]; unsigned int u; } p;
  p.h[0] = (__bf16)lo;
  p.h[1] = (__bf16)hi;
  return p.u;
}

__device__ __forceinline__ float gelu_tanh(float x) {
  float x3 = x * x * x;
  return 0.5f * x * (1.f + tanhf(0.7978845608028654f * (x + 0.044715f * x3)));
}

__device__ __forceinline__ float block_reduce_sum(float v, float* sred) {
  #pragma unroll
  for (int off = 16; off; off >>= 1) v += __shfl_xor(v, off, 32);
  int lane = threadIdx.x & 31, w = threadIdx.x >> 5;
  if (lane == 0) sred[w] = v;
  __syncthreads();
  float r = (lane < 8) ? sred[lane] : 0.f;
  #pragma unroll
  for (int off = 16; off; off >>= 1) r += __shfl_xor(r, off, 32);
  __syncthreads();
  return r;
}

// ---------------------------------------------------------------------------
// Generic strided/batched GEMM: C = epi(A[MxK] * B[KxN] (+bias)) (+Res)
// A,B f32 in global, converted to bf16, accum f32 via WMMA 16x16x32.
// Block: 256 threads = 8 waves (2x4), tile 64(M) x 128(N), K-step 32.
// LDS: A row-major K-contig; B staged n-major K-contig so BOTH operand
// fragments are two aligned ds_load_b128 per tile (no element repacks).
// flags: FLAG_BT -> B stored [N,K] row-major; FLAG_GELU -> tanh-gelu.
// Batched: z1 = z/bdiv, z2 = z%bdiv; offsets z1*s?1 + z2*s?2.
// ---------------------------------------------------------------------------
__global__ __launch_bounds__(256, 2)
void gemm_bf16_wmma(const float* __restrict__ A, const float* __restrict__ Bm,
                    float* __restrict__ C, const float* __restrict__ bias,
                    const float* __restrict__ Res,
                    int M, int N, int K, long lda, long ldb, long ldc,
                    int bdiv, long sA1, long sA2, long sB1, long sB2,
                    long sC1, long sC2, int flags) {
  __shared__ __bf16 As[64][40];     // 80B rows: 16B aligned, 20-bank stride
  __shared__ __bf16 Bst[128][40];   // n-major, K contiguous

  int z = blockIdx.z;
  int z1 = z / bdiv, z2 = z % bdiv;
  A  += (long)z1 * sA1 + (long)z2 * sA2;
  Bm += (long)z1 * sB1 + (long)z2 * sB2;
  long coff = (long)z1 * sC1 + (long)z2 * sC2;
  C += coff;
  if (Res) Res += coff;

  const int tid = threadIdx.x;
  const int lane = tid & 31, wave = tid >> 5;
  const int wm = wave & 1, wn = wave >> 1;   // 2x4 waves -> 64x128 tile
  const int row0 = blockIdx.y * 64;
  const int col0 = blockIdx.x * 128;
  const bool bt = (flags & FLAG_BT) != 0;
  const bool aFullM = (row0 + 64 <= M);
  const bool bFullN = (col0 + 128 <= N);

  float areg[8];
  float breg[16];

  auto load_a_tile = [&](int k0) {
    if (aFullM && (k0 + 32 <= K)) {
      #pragma unroll
      for (int j = 0; j < 4; ++j) {
        int p = tid + 256 * j;
        int r = p >> 4, c2 = p & 15;
        const float* src = A + (long)(row0 + r) * lda + (k0 + 2 * c2);
        areg[2 * j] = src[0];
        areg[2 * j + 1] = src[1];
      }
    } else {
      #pragma unroll
      for (int j = 0; j < 4; ++j) {
        int p = tid + 256 * j;
        int r = p >> 4, c2 = p & 15;
        int gr = row0 + r, gk = k0 + 2 * c2;
        const float* src = A + (long)gr * lda + gk;
        areg[2 * j]     = (gr < M && gk < K)     ? src[0] : 0.f;
        areg[2 * j + 1] = (gr < M && gk + 1 < K) ? src[1] : 0.f;
      }
    }
  };

  auto load_b_tile = [&](int k0) {
    if (bt) {
      if (bFullN && (k0 + 32 <= K)) {
        #pragma unroll
        for (int j = 0; j < 8; ++j) {
          int p = tid + 256 * j;
          int n = p >> 4, c2 = p & 15;
          const float* src = Bm + (long)(col0 + n) * ldb + (k0 + 2 * c2);
          breg[2 * j] = src[0];
          breg[2 * j + 1] = src[1];
        }
      } else {
        #pragma unroll
        for (int j = 0; j < 8; ++j) {
          int p = tid + 256 * j;
          int n = p >> 4, c2 = p & 15;
          int gn = col0 + n, gk = k0 + 2 * c2;
          const float* src = Bm + (long)gn * ldb + gk;
          breg[2 * j]     = (gn < N && gk < K)     ? src[0] : 0.f;
          breg[2 * j + 1] = (gn < N && gk + 1 < K) ? src[1] : 0.f;
        }
      }
    } else {
      if (bFullN && (k0 + 32 <= K)) {
        #pragma unroll
        for (int j = 0; j < 8; ++j) {
          int p = tid + 256 * j;
          int kk = p >> 6, c2 = p & 63;
          const float* src = Bm + (long)(k0 + kk) * ldb + (col0 + 2 * c2);
          breg[2 * j] = src[0];
          breg[2 * j + 1] = src[1];
        }
      } else {
        #pragma unroll
        for (int j = 0; j < 8; ++j) {
          int p = tid + 256 * j;
          int kk = p >> 6, c2 = p & 63;
          int gk = k0 + kk, gn = col0 + 2 * c2;
          const float* src = Bm + (long)gk * ldb + gn;
          breg[2 * j]     = (gk < K && gn < N)     ? src[0] : 0.f;
          breg[2 * j + 1] = (gk < K && gn + 1 < N) ? src[1] : 0.f;
        }
      }
    }
  };

  auto store_tiles = [&]() {
    #pragma unroll
    for (int j = 0; j < 4; ++j) {
      int p = tid + 256 * j;
      int r = p >> 4, c2 = p & 15;
      *(unsigned int*)&As[r][2 * c2] = pack_bf16(areg[2 * j], areg[2 * j + 1]);
    }
    if (bt) {
      #pragma unroll
      for (int j = 0; j < 8; ++j) {
        int p = tid + 256 * j;
        int n = p >> 4, c2 = p & 15;
        *(unsigned int*)&Bst[n][2 * c2] = pack_bf16(breg[2 * j], breg[2 * j + 1]);
      }
    } else {
      #pragma unroll
      for (int j = 0; j < 8; ++j) {
        int p = tid + 256 * j;
        int kk = p >> 6, c2 = p & 63;
        int n = 2 * c2;
        Bst[n][kk]     = (__bf16)breg[2 * j];
        Bst[n + 1][kk] = (__bf16)breg[2 * j + 1];
      }
    }
  };

  v8f acc[2][2] = {};
  const int mrow = lane & 15;
  const int kga = (lane >> 4) * 8;    // A: ISA 16-bit 16x32 lane layout
  const int kgb = (lane >> 4) * 8;    // B: same ISA 16-bit lane layout as A

  load_a_tile(0);
  load_b_tile(0);
  for (int k0 = 0; k0 < K; k0 += 32) {
    store_tiles();
    __syncthreads();
    int kn = k0 + 32;
    if (kn < K) {            // issue next tile's loads under the WMMAs
      load_a_tile(kn);
      load_b_tile(kn);
    }

    fragu fa[2], fb[2];
    #pragma unroll
    for (int tm = 0; tm < 2; ++tm) {
      int r = wm * 32 + tm * 16 + mrow;
      fa[tm].q[0] = *(const v4u*)&As[r][kga];
      fa[tm].q[1] = *(const v4u*)&As[r][kga + 16];
    }
    #pragma unroll
    for (int tn = 0; tn < 2; ++tn) {
      int c = wn * 32 + tn * 16 + mrow;
      fb[tn].q[0] = *(const v4u*)&Bst[c][kgb];
      fb[tn].q[1] = *(const v4u*)&Bst[c][kgb + 16];
    }
    #pragma unroll
    for (int tm = 0; tm < 2; ++tm)
      #pragma unroll
      for (int tn = 0; tn < 2; ++tn)
        acc[tm][tn] = __builtin_amdgcn_wmma_f32_16x16x32_bf16(
            false, fa[tm].v, false, fb[tn].v, (short)0, acc[tm][tn], false, false);
    __syncthreads();
  }

  // Epilogue. C/D layout: lanes 0-15 -> M=vgpr r, lanes 16-31 -> M=8+r; N=lane&15.
  int nlane = lane & 15;
  int mhi = (lane >> 4) * 8;
  bool do_gelu = (flags & FLAG_GELU) != 0;
  #pragma unroll
  for (int tm = 0; tm < 2; ++tm)
    #pragma unroll
    for (int tn = 0; tn < 2; ++tn) {
      int gcol = col0 + wn * 32 + tn * 16 + nlane;
      #pragma unroll
      for (int r = 0; r < 8; ++r) {
        int grow = row0 + wm * 32 + tm * 16 + mhi + r;
        if (grow < M && gcol < N) {
          float v = acc[tm][tn][r];
          if (bias) v += bias[gcol];
          if (do_gelu) v = gelu_tanh(v);
          if (Res) v += Res[(long)grow * ldc + gcol];
          C[(long)grow * ldc + gcol] = v;
        }
      }
    }
}

// ---------------------------------------------------------------------------
// LayerNorm over rows of 768. Row addressing:
//   off = base + (row / rows_per_b)*stride_b + (row % rows_per_b)*stride_r
// ---------------------------------------------------------------------------
__global__ __launch_bounds__(256)
void layernorm_k(const float* __restrict__ X, float* __restrict__ Y,
                 const float* __restrict__ gs, const float* __restrict__ gb,
                 long base, long stride_b, int rows_per_b, long stride_r,
                 long out_stride) {
  __shared__ float sred[8];
  int row = blockIdx.x;
  long off = base + (long)(row / rows_per_b) * stride_b +
             (long)(row % rows_per_b) * stride_r;
  const float* x = X + off;
  float* y = Y + (long)row * out_stride;
  int t = threadIdx.x;
  float a = x[t], b = x[t + 256], c = x[t + 512];
  float s = block_reduce_sum(a + b + c, sred);
  float mu = s * (1.f / 768.f);
  float da = a - mu, db = b - mu, dc = c - mu;
  float vs = block_reduce_sum(da * da + db * db + dc * dc, sred);
  float rstd = rsqrtf(vs * (1.f / 768.f) + 1e-6f);
  y[t]       = da * rstd * gs[t]       + gb[t];
  y[t + 256] = db * rstd * gs[t + 256] + gb[t + 256];
  y[t + 512] = dc * rstd * gs[t + 512] + gb[t + 512];
}

// Softmax over rows of length S (<=222), one wave per row, scale applied first.
__global__ __launch_bounds__(256)
void softmax_k(float* __restrict__ X, int rows, int S, float scale) {
  int w = threadIdx.x >> 5, lane = threadIdx.x & 31;
  int row = blockIdx.x * 8 + w;
  if (row >= rows) return;
  float* x = X + (long)row * S;
  float m = -1e30f;
  for (int i = lane; i < S; i += 32) m = fmaxf(m, x[i] * scale);
  #pragma unroll
  for (int off = 16; off; off >>= 1) m = fmaxf(m, __shfl_xor(m, off, 32));
  float e[7];
  float s = 0.f;
  int c = 0;
  for (int i = lane; i < S; i += 32) {
    float v = __expf(x[i] * scale - m);
    e[c++] = v;
    s += v;
  }
  #pragma unroll
  for (int off = 16; off; off >>= 1) s += __shfl_xor(s, off, 32);
  float inv = 1.f / s;
  c = 0;
  for (int i = lane; i < S; i += 32) x[i] = e[c++] * inv;
}

// im2col: [B,3,224,224] -> [B*196, 768] with feature = c*256 + py*16 + px.
__global__ void im2col_k(const float* __restrict__ inp, float* __restrict__ xp) {
  long idx = (long)blockIdx.x * blockDim.x + threadIdx.x;
  long total = (long)NB_ * 196 * 768;
  if (idx >= total) return;
  int f = (int)(idx % 768);
  long r = idx / 768;
  int p = (int)(r % 196);
  int b = (int)(r / 196);
  int c = f >> 8, q = f & 255, py = q >> 4, px = q & 15;
  int pr = p / 14, pc = p % 14;
  xp[idx] = inp[(((long)b * 3 + c) * 224 + pr * 16 + py) * 224 + pc * 16 + px];
}

// tokens + pos: tpe[b,0]=cls+pos0 ; tpe[b,1+p]=xproj[b,p]+pos[1+p]
__global__ void build_tokens_k(const float* __restrict__ xproj,
                               const float* __restrict__ cls,
                               const float* __restrict__ pos,
                               float* __restrict__ tpe) {
  long idx = (long)blockIdx.x * blockDim.x + threadIdx.x;
  long total = (long)NB_ * S1 * 768;
  if (idx >= total) return;
  int f = (int)(idx % 768);
  long r = idx / 768;
  int s = (int)(r % S1);
  int b = (int)(r / S1);
  float v = (s == 0) ? cls[f] : xproj[((long)b * 196 + (s - 1)) * 768 + f];
  tpe[idx] = v + pos[(long)s * 768 + f];
}

// sx = concat(tpe[:, :1], prompts(topk)+pos0, tpe[:, 1:])
__global__ void build_prompted_k(const float* __restrict__ tpe,
                                 const float* __restrict__ prompt,
                                 const float* __restrict__ pos,
                                 const int* __restrict__ topk,
                                 float* __restrict__ sx) {
  long idx = (long)blockIdx.x * blockDim.x + threadIdx.x;
  long total = (long)NB_ * S2 * 768;
  if (idx >= total) return;
  int f = (int)(idx % 768);
  long r = idx / 768;
  int s = (int)(r % S2);
  int b = (int)(r / S2);
  float v;
  if (s == 0) {
    v = tpe[((long)b * S1) * 768 + f];
  } else if (s <= SELN * PLENN) {
    int i = (s - 1) / PLENN, j = (s - 1) % PLENN;
    int pp = topk[b * SELN + i];
    v = prompt[((long)pp * PLENN + j) * 768 + f] + pos[f];
  } else {
    v = tpe[((long)b * S1 + (s - SELN * PLENN)) * 768 + f];
  }
  sx[idx] = v;
}

// ---------------- Blocked Cholesky (lower), NB=64, 10 matrices -------------
__global__ __launch_bounds__(256)
void chol_panel_k(float* __restrict__ Ach, int kb) {
  float* Ap = Ach + (long)blockIdx.x * 768 * 768;
  __shared__ float dinv;
  int j0 = kb * 64;
  int tid = threadIdx.x;
  for (int jj = 0; jj < 64; ++jj) {
    int j = j0 + jj;
    if (tid == 0) {
      float d = sqrtf(fmaxf(Ap[(long)j * 768 + j], 1e-20f));
      Ap[(long)j * 768 + j] = d;
      dinv = 1.f / d;
    }
    __syncthreads();
    for (int i = j + 1 + tid; i < 768; i += 256) Ap[(long)i * 768 + j] *= dinv;
    __syncthreads();
    int ncols = 63 - jj;
    for (int t = tid; t < ncols * 768; t += 256) {
      int jc = jj + 1 + t / 768;
      int i = t % 768;
      int colj = j0 + jc;
      if (i >= colj)
        Ap[(long)i * 768 + colj] -= Ap[(long)i * 768 + j] * Ap[(long)colj * 768 + j];
    }
    __syncthreads();
  }
}

__global__ __launch_bounds__(256)
void chol_syrk_k(float* __restrict__ Ach, int kb) {
  float* Ap = Ach + (long)blockIdx.z * 768 * 768;
  int e = (kb + 1) * 64, j0 = kb * 64;
  int ti = blockIdx.y, tj = blockIdx.x;
  if (tj > ti) return;
  int r0 = e + ti * 32, c0 = e + tj * 32;
  if (r0 >= 768) return;
  __shared__ float Li[32][65];
  __shared__ float Lj[32][65];
  int tid = threadIdx.x;
  for (int t = tid; t < 32 * 64; t += 256) {
    int r = t >> 6, k = t & 63;
    Li[r][k] = (r0 + r < 768) ? Ap[(long)(r0 + r) * 768 + j0 + k] : 0.f;
  }
  for (int t = tid; t < 32 * 64; t += 256) {
    int r = t >> 6, k = t & 63;
    Lj[r][k] = (c0 + r < 768) ? Ap[(long)(c0 + r) * 768 + j0 + k] : 0.f;
  }
  __syncthreads();
  int r = tid >> 3;
  int cb = (tid & 7) * 4;
  float acc[4] = {0.f, 0.f, 0.f, 0.f};
  for (int k = 0; k < 64; ++k) {
    float a = Li[r][k];
    #pragma unroll
    for (int cc = 0; cc < 4; ++cc) acc[cc] += a * Lj[cb + cc][k];
  }
  #pragma unroll
  for (int cc = 0; cc < 4; ++cc) {
    int gr = r0 + r, gc = c0 + cb + cc;
    if (gr < 768 && gc < 768 && gc <= gr) Ap[(long)gr * 768 + gc] -= acc[cc];
  }
}

__global__ __launch_bounds__(256)
void logdet_k(const float* __restrict__ Ach, float* __restrict__ ld) {
  __shared__ float sred[8];
  const float* Ap = Ach + (long)blockIdx.x * 768 * 768;
  float s = 0.f;
  for (int i = threadIdx.x; i < 768; i += 256)
    s += logf(fmaxf(Ap[(long)i * 768 + i], 1e-30f));
  float tot = block_reduce_sum(s, sred);
  if (threadIdx.x == 0) ld[blockIdx.x] = 2.f * tot;
}

// maha[b,p] = || L_p^{-1} (q_b - mean_p) ||^2  via forward substitution.
__global__ __launch_bounds__(256)
void forsub_maha_k(const float* __restrict__ Lmat, const float* __restrict__ qv,
                   const float* __restrict__ meanv, float* __restrict__ maha) {
  __shared__ float y[768];
  __shared__ float sred[8];
  int b = blockIdx.x / POOLN, p = blockIdx.x % POOLN;
  const float* L = Lmat + (long)p * 768 * 768;
  int tid = threadIdx.x;
  for (int i = tid; i < 768; i += 256) y[i] = qv[b * 768 + i] - meanv[p * 768 + i];
  __syncthreads();
  for (int i = 0; i < 768; ++i) {
    float part = 0.f;
    for (int j = tid; j < i; j += 256) part += L[(long)i * 768 + j] * y[j];
    float dot = block_reduce_sum(part, sred);
    if (tid == 0) y[i] = (y[i] - dot) / L[(long)i * 768 + i];
    __syncthreads();
  }
  float part = 0.f;
  for (int j = tid; j < 768; j += 256) part += y[j] * y[j];
  float s = block_reduce_sum(part, sred);
  if (tid == 0) maha[b * POOLN + p] = s;
}

__global__ void score_topk_k(const float* __restrict__ maha,
                             const float* __restrict__ ld,
                             const float* __restrict__ freq,
                             int* __restrict__ topk) {
  int b = threadIdx.x;
  if (b >= NB_) return;
  float fsum = 0.f;
  for (int p = 0; p < POOLN; ++p) fsum += freq[p];
  const float LOG2PI = 1.8378770664093453f;
  float sc[POOLN];
  for (int p = 0; p < POOLN; ++p) {
    float lp = -0.5f * (maha[b * POOLN + p] + ld[p] + 768.f * LOG2PI);
    sc[p] = lp * (freq[p] / fsum);
  }
  bool used[POOLN] = {};
  for (int s = 0; s < SELN; ++s) {
    int best = 0;
    float bv = -3.4e38f;
    for (int p = 0; p < POOLN; ++p)
      if (!used[p] && sc[p] > bv) { bv = sc[p]; best = p; }
    used[best] = true;
    topk[b * SELN + s] = best;
  }
}

__global__ void mean25_k(const float* __restrict__ P, float* __restrict__ G) {
  int idx = blockIdx.x * blockDim.x + threadIdx.x;
  if (idx >= NB_ * 768) return;
  int b = idx / 768, f = idx % 768;
  float s = 0.f;
  for (int j = 0; j < SELN * PLENN; ++j)
    s += P[((long)b * SELN * PLENN + j) * 768 + f];
  G[idx] = s * (1.f / (SELN * PLENN));
}

// ---------------------------------------------------------------------------
// Host side
// ---------------------------------------------------------------------------
static inline void launch_gemm(hipStream_t st, const float* A, const float* Bm,
                               float* C, const float* bias, const float* Res,
                               int M, int N, int K, long lda, long ldb, long ldc,
                               int flags, int nbatch, int bdiv, long sA1, long sA2,
                               long sB1, long sB2, long sC1, long sC2) {
  dim3 g((N + 127) / 128, (M + 63) / 64, nbatch);
  gemm_bf16_wmma<<<g, 256, 0, st>>>(A, Bm, C, bias, Res, M, N, K, lda, ldb, ldc,
                                    bdiv, sA1, sA2, sB1, sB2, sC1, sC2, flags);
}

static void run_block(hipStream_t st, float* xb, int S, int l,
                      const float* ln1_s, const float* ln1_b,
                      const float* qkv_w, const float* qkv_b,
                      const float* proj_w, const float* proj_b,
                      const float* ln2_s, const float* ln2_b,
                      const float* fc1_w, const float* fc1_b,
                      const float* fc2_w, const float* fc2_b,
                      float* hbuf, float* qkvbuf, float* attbuf, float* obuf,
                      float* mbuf) {
  const int M = NB_ * S;
  const long D = 768;
  const float* l1s = ln1_s + (long)l * D;
  const float* l1b = ln1_b + (long)l * D;
  const float* qw = qkv_w + (long)l * D * 3 * D;
  const float* qb = qkv_b + (long)l * 3 * D;
  const float* pw = proj_w + (long)l * D * D;
  const float* pb = proj_b + (long)l * D;
  const float* l2s = ln2_s + (long)l * D;
  const float* l2b = ln2_b + (long)l * D;
  const float* f1w = fc1_w + (long)l * D * 4 * D;
  const float* f1b = fc1_b + (long)l * 4 * D;
  const float* f2w = fc2_w + (long)l * 4 * D * D;
  const float* f2b = fc2_b + (long)l * D;

  // h = LN1(x)
  layernorm_k<<<M, 256, 0, st>>>(xb, hbuf, l1s, l1b, 0, 0, M, D, D);
  // qkv = h @ qkv_w + qkv_b
  launch_gemm(st, hbuf, qw, qkvbuf, qb, nullptr, M, 3 * 768, 768, D, 3 * D, 3 * D,
              0, 1, 1, 0, 0, 0, 0, 0, 0);
  // att = q @ k^T (per b,h), B-transposed mode
  launch_gemm(st, qkvbuf, qkvbuf + 768, attbuf, nullptr, nullptr, S, S, DHEAD,
              3 * D, 3 * D, S, FLAG_BT, NB_ * NHEAD, NHEAD,
              (long)S * 3 * D, DHEAD, (long)S * 3 * D, DHEAD,
              (long)NHEAD * S * S, (long)S * S);
  // softmax(att / 8)
  int rows = NB_ * NHEAD * S;
  softmax_k<<<(rows + 7) / 8, 256, 0, st>>>(attbuf, rows, S, 0.125f);
  // o = att @ v
  launch_gemm(st, attbuf, qkvbuf + 1536, obuf, nullptr, nullptr, S, DHEAD, S,
              S, 3 * D, D, 0, NB_ * NHEAD, NHEAD,
              (long)NHEAD * S * S, (long)S * S, (long)S * 3 * D, DHEAD,
              (long)S * D, DHEAD);
  // x = x + o @ proj_w + proj_b (in-place residual)
  launch_gemm(st, obuf, pw, xb, pb, xb, M, 768, 768, D, D, D, 0, 1, 1,
              0, 0, 0, 0, 0, 0);
  // h = LN2(x)
  layernorm_k<<<M, 256, 0, st>>>(xb, hbuf, l2s, l2b, 0, 0, M, D, D);
  // m = gelu(h @ fc1_w + fc1_b)
  launch_gemm(st, hbuf, f1w, mbuf, f1b, nullptr, M, 4 * 768, 768, D, 4 * D, 4 * D,
              FLAG_GELU, 1, 1, 0, 0, 0, 0, 0, 0);
  // x = x + m @ fc2_w + fc2_b
  launch_gemm(st, mbuf, f2w, xb, f2b, xb, M, 768, 4 * 768, 4 * D, D, D, 0, 1, 1,
              0, 0, 0, 0, 0, 0);
}

extern "C" void kernel_launch(void* const* d_in, const int* in_sizes, int n_in,
                              void* d_out, int out_size, void* d_ws, size_t ws_size,
                              hipStream_t stream) {
  (void)in_sizes; (void)n_in; (void)out_size; (void)ws_size;
  const float* inp      = (const float*)d_in[0];
  const float* patch_w  = (const float*)d_in[1];
  const float* patch_b  = (const float*)d_in[2];
  const float* cls_tok  = (const float*)d_in[3];
  const float* pos_emb  = (const float*)d_in[4];
  const float* ln1_s    = (const float*)d_in[5];
  const float* ln1_b    = (const float*)d_in[6];
  const float* qkv_w    = (const float*)d_in[7];
  const float* qkv_b    = (const float*)d_in[8];
  const float* proj_w   = (const float*)d_in[9];
  const float* proj_b   = (const float*)d_in[10];
  const float* ln2_s    = (const float*)d_in[11];
  const float* ln2_b    = (const float*)d_in[12];
  const float* fc1_w    = (const float*)d_in[13];
  const float* fc1_b    = (const float*)d_in[14];
  const float* fc2_w    = (const float*)d_in[15];
  const float* fc2_b    = (const float*)d_in[16];
  const float* norm_s   = (const float*)d_in[17];
  const float* norm_b   = (const float*)d_in[18];
  const float* head_w   = (const float*)d_in[19];
  const float* head_b   = (const float*)d_in[20];
  const float* prompt   = (const float*)d_in[21];
  const float* meanv    = (const float*)d_in[22];
  const float* variance = (const float*)d_in[23];
  const float* freq     = (const float*)d_in[24];
  float* out = (float*)d_out;

  float* W = (float*)d_ws;
  size_t o = 0;
  auto carve = [&](size_t n) { size_t r = o; o += (n + 63) & ~(size_t)63; return r; };
  const long D = 768;
  float* tpe   = W + carve((size_t)NB_ * S1 * D);            // tokens + pos
  float* xbuf  = W + carve((size_t)NB_ * S2 * D);            // residual stream
  float* qkvb  = W + carve((size_t)NB_ * S2 * 3 * D);        // qkv
  float* attb  = W + carve((size_t)NB_ * NHEAD * S2 * S2);   // att (aliases h)
  float* mb    = W + carve((size_t)NB_ * S2 * 4 * D);        // mlp hidden (aliases o, im2col)
  float* chol  = W + carve((size_t)POOLN * D * D);           // cholesky workspace
  float* qvec  = W + carve((size_t)NB_ * D);
  float* pbuf  = W + carve((size_t)NB_ * SELN * PLENN * D);
  float* gvec  = W + carve((size_t)NB_ * D);
  float* maha  = W + carve((size_t)NB_ * POOLN);
  float* ldet  = W + carve((size_t)16);
  int*   topk  = (int*)(W + carve((size_t)NB_ * SELN));
  float* hbuf  = attb;   // LN output; lifetime disjoint from att
  float* obuf  = mb;     // attn output; lifetime disjoint from mlp hidden
  float* xp    = mb;     // im2col buffer; used before layers only
  float* xproj = attb;   // patch projection; used before layers only

  // ---- patch embed ----
  {
    long tot = (long)NB_ * 196 * D;
    im2col_k<<<(int)((tot + 255) / 256), 256, 0, stream>>>(inp, xp);
    launch_gemm(stream, xp, patch_w, xproj, patch_b, nullptr,
                NB_ * 196, 768, 768, D, D, D, 0, 1, 1, 0, 0, 0, 0, 0, 0);
    long tot2 = (long)NB_ * S1 * D;
    build_tokens_k<<<(int)((tot2 + 255) / 256), 256, 0, stream>>>(xproj, cls_tok,
                                                                  pos_emb, tpe);
  }

  // ---- pass 1: plain ViT on [B,197,D] ----
  hipMemcpyAsync(xbuf, tpe, (size_t)NB_ * S1 * D * sizeof(float),
                 hipMemcpyDeviceToDevice, stream);
  for (int l = 0; l < 12; ++l)
    run_block(stream, xbuf, S1, l, ln1_s, ln1_b, qkv_w, qkv_b, proj_w, proj_b,
              ln2_s, ln2_b, fc1_w, fc1_b, fc2_w, fc2_b, hbuf, qkvb, attb, obuf, mb);
  // q = LN(x)[:,0]
  layernorm_k<<<NB_, 256, 0, stream>>>(xbuf, qvec, norm_s, norm_b,
                                       0, (long)S1 * D, 1, 0, D);

  // ---- Gaussian scoring: Cholesky of variance, maha via forward-subst ----
  hipMemcpyAsync(chol, variance, (size_t)POOLN * D * D * sizeof(float),
                 hipMemcpyDeviceToDevice, stream);
  for (int kb = 0; kb < 12; ++kb) {
    chol_panel_k<<<POOLN, 256, 0, stream>>>(chol, kb);
    int e = (kb + 1) * 64;
    int T = 768 - e;
    if (T > 0) {
      int tiles = (T + 31) / 32;
      dim3 g(tiles, tiles, POOLN);
      chol_syrk_k<<<g, 256, 0, stream>>>(chol, kb);
    }
  }
  logdet_k<<<POOLN, 256, 0, stream>>>(chol, ldet);
  forsub_maha_k<<<NB_ * POOLN, 256, 0, stream>>>(chol, qvec, meanv, maha);
  score_topk_k<<<1, 32, 0, stream>>>(maha, ldet, freq, topk);

  // ---- pass 2: prompted sequence [B,222,D] ----
  {
    long tot = (long)NB_ * S2 * D;
    build_prompted_k<<<(int)((tot + 255) / 256), 256, 0, stream>>>(tpe, prompt,
                                                                   pos_emb, topk,
                                                                   xbuf);
  }
  for (int l = 0; l < 12; ++l)
    run_block(stream, xbuf, S2, l, ln1_s, ln1_b, qkv_w, qkv_b, proj_w, proj_b,
              ln2_s, ln2_b, fc1_w, fc1_b, fc2_w, fc2_b, hbuf, qkvb, attb, obuf, mb);

  // LN of prompt rows 1..25, mean, head
  layernorm_k<<<NB_ * SELN * PLENN, 256, 0, stream>>>(
      xbuf, pbuf, norm_s, norm_b, D, (long)S2 * D, SELN * PLENN, D, D);
  mean25_k<<<(NB_ * 768 + 255) / 256, 256, 0, stream>>>(pbuf, gvec);
  launch_gemm(stream, gvec, head_w, out, head_b, nullptr, NB_, 100, 768,
              D, 100, 100, 0, 1, 1, 0, 0, 0, 0, 0, 0);
}
